// SinkhornRouter_57423712747980
// MI455X (gfx1250) — matrix operand order repaired and checked by reading, not compiled, not run on hardware
//
#include <hip/hip_runtime.h>

// Problem constants (reference): b=4, n=4096, d=1024, e=8, h=1, m=512
#define BB 4
#define NN 4096
#define DD 1024
#define EE 8
#define MM 512
#define LDA 40   // padded LDS row pitch in halves: 80B -> 16B-aligned, bank-conflict-free

// ---- gfx1250 async global->LDS support (guarded) --------------------------
#if defined(__has_builtin)
# if __has_builtin(__builtin_amdgcn_global_load_async_to_lds_b128)
#  define HAVE_ASYNC_LDS 1
# endif
# if __has_builtin(__builtin_amdgcn_s_wait_asynccnt)
#  define HAVE_WAIT_ASYNC_BUILTIN 1
# endif
#endif
#ifndef HAVE_ASYNC_LDS
# define HAVE_ASYNC_LDS 0
#endif
#ifndef HAVE_WAIT_ASYNC_BUILTIN
# define HAVE_WAIT_ASYNC_BUILTIN 0
#endif

typedef __attribute__((ext_vector_type(16))) __bf16 v16bf;
typedef __attribute__((ext_vector_type(8)))  float  v8f;
typedef int v4i __attribute__((ext_vector_type(4)));

union V16 { uint4 q[2]; v16bf v; };

__device__ __forceinline__ unsigned short f2bf(float f) {
    unsigned int u = __float_as_uint(f);
    unsigned int r = u + 0x7FFFu + ((u >> 16) & 1u);   // round-to-nearest-even
    return (unsigned short)(r >> 16);
}

#if HAVE_ASYNC_LDS
typedef v4i __attribute__((address_space(1))) *as1_v4i;
typedef v4i __attribute__((address_space(3))) *as3_v4i;

// copy 32 contiguous bytes global -> LDS via two async b128 ops (ASYNCcnt)
__device__ __forceinline__ void async_cp32(const unsigned short* g, unsigned ldsByte) {
    __builtin_amdgcn_global_load_async_to_lds_b128(
        (as1_v4i)(uintptr_t)g,
        (as3_v4i)(uintptr_t)ldsByte, 0, 0);
    __builtin_amdgcn_global_load_async_to_lds_b128(
        (as1_v4i)((uintptr_t)g + 16),
        (as3_v4i)(uintptr_t)(ldsByte + 16), 0, 0);
}
#endif

__device__ __forceinline__ void wait_async0() {
#if HAVE_WAIT_ASYNC_BUILTIN
    __builtin_amdgcn_s_wait_asynccnt(0);
#else
    asm volatile("s_wait_asynccnt 0x0" ::: "memory");
#endif
}

// ---------------------------------------------------------------------------
// K0a: bulk fp32 -> bf16 convert (x).  Bandwidth-bound; 8 elements/thread.
// ---------------------------------------------------------------------------
__global__ __launch_bounds__(256) void k_cvt_bf16(
    const float* __restrict__ in, unsigned short* __restrict__ out)
{
    size_t i = (size_t)blockIdx.x * 256 + threadIdx.x;   // handles 8 elements
    const float4* p = (const float4*)in + i * 2;
    float4 a = p[0], b = p[1];
    union { unsigned short h[8]; uint4 q; } u;
    u.h[0] = f2bf(a.x); u.h[1] = f2bf(a.y); u.h[2] = f2bf(a.z); u.h[3] = f2bf(a.w);
    u.h[4] = f2bf(b.x); u.h[5] = f2bf(b.y); u.h[6] = f2bf(b.z); u.h[7] = f2bf(b.w);
    *((uint4*)(out + i * 8)) = u.q;
}

// ---------------------------------------------------------------------------
// K0b: experts fp32 [e][k][n]  ->  bf16 transposed [e][n][k] (32x32 LDS tiles)
// ---------------------------------------------------------------------------
__global__ __launch_bounds__(256) void k_cvt_transpose(
    const float* __restrict__ experts, unsigned short* __restrict__ outT)
{
    __shared__ unsigned short t[32][34];
    int bid = blockIdx.x;
    int e  = bid >> 10;
    int kt = (bid >> 5) & 31;
    int nt = bid & 31;
    int tid = threadIdx.x;

    int r  = tid >> 3;            // k within tile, 0..31
    int c4 = (tid & 7) * 4;       // n within tile
    float4 v = *(const float4*)(experts + ((size_t)e * DD + kt * 32 + r) * DD + nt * 32 + c4);
    t[c4 + 0][r] = f2bf(v.x);
    t[c4 + 1][r] = f2bf(v.y);
    t[c4 + 2][r] = f2bf(v.z);
    t[c4 + 3][r] = f2bf(v.w);
    __syncthreads();

    int n  = tid >> 3;            // n within tile
    int k4 = (tid & 7) * 4;       // k within tile
    union { unsigned short h[4]; uint2 q; } u;
    u.h[0] = t[n][k4 + 0]; u.h[1] = t[n][k4 + 1];
    u.h[2] = t[n][k4 + 2]; u.h[3] = t[n][k4 + 3];
    *((uint2*)(outT + ((size_t)e * DD + nt * 32 + n) * DD + kt * 32 + k4)) = u.q;
}

// ---------------------------------------------------------------------------
// K1: gate logits t = log(clip(x @ gate_w, 1e-6)).  One wave per token row.
// ---------------------------------------------------------------------------
__global__ __launch_bounds__(256) void k_gate_logits(
    const float* __restrict__ x, const float* __restrict__ gw, float* __restrict__ t)
{
    __shared__ float gws[DD * EE];   // 32 KB
    int tid = threadIdx.x;
    for (int i = tid; i < DD * EE; i += 256) gws[i] = gw[i];
    __syncthreads();

    int wid = tid >> 5, lane = tid & 31;
    int row = blockIdx.x * 8 + wid;          // 0..16383  (b*n flattened)
    const float* xr = x + (size_t)row * DD;

    float acc[EE];
    #pragma unroll
    for (int j = 0; j < EE; ++j) acc[j] = 0.0f;

    for (int i = lane; i < DD; i += 32) {
        float xv = xr[i];
        const float* g = &gws[i * EE];
        #pragma unroll
        for (int j = 0; j < EE; ++j) acc[j] = fmaf(xv, g[j], acc[j]);
    }
    #pragma unroll
    for (int j = 0; j < EE; ++j) {
        #pragma unroll
        for (int off = 16; off >= 1; off >>= 1)
            acc[j] += __shfl_xor(acc[j], off, 32);
    }
    if (lane == 0) {
        float* tr = t + (size_t)row * EE;
        #pragma unroll
        for (int j = 0; j < EE; ++j)
            tr[j] = logf(fmaxf(acc[j], 1e-6f));
    }
}

// ---------------------------------------------------------------------------
// K2: Sinkhorn, one block per batch, rows held in registers.
// ---------------------------------------------------------------------------
__global__ __launch_bounds__(1024) void k_sinkhorn(float* __restrict__ t)
{
    int b = blockIdx.x;
    int tid = threadIdx.x;
    int wid = tid >> 5, lane = tid & 31;
    float* tb = t + (size_t)b * NN * EE;

    float v[4][EE];
    #pragma unroll
    for (int q = 0; q < 4; ++q) {
        const float4* p = (const float4*)&tb[(size_t)(tid * 4 + q) * EE];
        float4 a = p[0], c = p[1];
        v[q][0] = a.x; v[q][1] = a.y; v[q][2] = a.z; v[q][3] = a.w;
        v[q][4] = c.x; v[q][5] = c.y; v[q][6] = c.z; v[q][7] = c.w;
    }

    __shared__ float wred[32 * EE];
    __shared__ float cmaxs[EE];
    __shared__ float clses[EE];

    for (int it = 0; it < 8; ++it) {
        float cm[EE];
        #pragma unroll
        for (int j = 0; j < EE; ++j) {
            float m = v[0][j];
            #pragma unroll
            for (int q = 1; q < 4; ++q) m = fmaxf(m, v[q][j]);
            #pragma unroll
            for (int off = 16; off >= 1; off >>= 1)
                m = fmaxf(m, __shfl_xor(m, off, 32));
            cm[j] = m;
        }
        if (lane == 0) {
            #pragma unroll
            for (int j = 0; j < EE; ++j) wred[wid * EE + j] = cm[j];
        }
        __syncthreads();
        if (tid < EE) {
            float m = wred[tid];
            for (int w = 1; w < 32; ++w) m = fmaxf(m, wred[w * EE + tid]);
            cmaxs[tid] = m;
        }
        __syncthreads();
        #pragma unroll
        for (int j = 0; j < EE; ++j) cm[j] = cmaxs[j];
        __syncthreads();

        float cs[EE];
        #pragma unroll
        for (int j = 0; j < EE; ++j) {
            float s = 0.0f;
            #pragma unroll
            for (int q = 0; q < 4; ++q) s += __expf(v[q][j] - cm[j]);
            #pragma unroll
            for (int off = 16; off >= 1; off >>= 1)
                s += __shfl_xor(s, off, 32);
            cs[j] = s;
        }
        if (lane == 0) {
            #pragma unroll
            for (int j = 0; j < EE; ++j) wred[wid * EE + j] = cs[j];
        }
        __syncthreads();
        if (tid < EE) {
            float s = 0.0f;
            for (int w = 0; w < 32; ++w) s += wred[w * EE + tid];
            clses[tid] = cmaxs[tid] + logf(s);
        }
        __syncthreads();
        float cl[EE];
        #pragma unroll
        for (int j = 0; j < EE; ++j) cl[j] = clses[j];

        #pragma unroll
        for (int q = 0; q < 4; ++q) {
            float rm = -3.4e38f;
            #pragma unroll
            for (int j = 0; j < EE; ++j) { v[q][j] -= cl[j]; rm = fmaxf(rm, v[q][j]); }
            float rs = 0.0f;
            #pragma unroll
            for (int j = 0; j < EE; ++j) rs += __expf(v[q][j] - rm);
            float rl = rm + logf(rs);
            #pragma unroll
            for (int j = 0; j < EE; ++j) v[q][j] -= rl;
        }
        __syncthreads();
    }

    #pragma unroll
    for (int q = 0; q < 4; ++q) {
        float4 a, c;
        a.x = __expf(v[q][0]); a.y = __expf(v[q][1]); a.z = __expf(v[q][2]); a.w = __expf(v[q][3]);
        c.x = __expf(v[q][4]); c.y = __expf(v[q][5]); c.z = __expf(v[q][6]); c.w = __expf(v[q][7]);
        float4* p = (float4*)&tb[(size_t)(tid * 4 + q) * EE];
        p[0] = a; p[1] = c;
    }
}

// ---------------------------------------------------------------------------
// K3: per-(b,e) exact top-512 via bitonic sort + winner atomicMax.
// ---------------------------------------------------------------------------
__global__ __launch_bounds__(1024) void k_topk(
    const float* __restrict__ gates, float* __restrict__ tv, int* __restrict__ ti,
    int* __restrict__ winner)
{
    int be = blockIdx.x;
    int b = be >> 3, e = be & 7;
    __shared__ float val[NN];
    __shared__ int   idx[NN];
    int tid = threadIdx.x;

    const float* g = gates + (size_t)b * NN * EE;
    for (int i = tid; i < NN; i += 1024) { val[i] = g[(size_t)i * EE + e]; idx[i] = i; }
    __syncthreads();

    for (int k = 2; k <= NN; k <<= 1) {
        for (int j = k >> 1; j > 0; j >>= 1) {
            #pragma unroll
            for (int s = 0; s < 2; ++s) {
                int i = tid + s * 1024;
                int ixj = i ^ j;
                if (ixj > i) {
                    bool up = ((i & k) == 0);
                    float va = val[i], vb = val[ixj];
                    int ia = idx[i], ib = idx[ixj];
                    bool aFirst = (va > vb) || (va == vb && ia < ib);
                    if (aFirst != up) {
                        val[i] = vb; val[ixj] = va;
                        idx[i] = ib; idx[ixj] = ia;
                    }
                }
            }
            __syncthreads();
        }
    }

    if (tid < MM) {
        tv[(size_t)be * MM + tid] = val[tid];
        ti[(size_t)be * MM + tid] = idx[tid];
        atomicMax(&winner[b * NN + idx[tid]], tid * EE + e);
    }
}

// ---------------------------------------------------------------------------
// K4: per-(b,e) expert GEMM on WMMA bf16.  Double-buffered LDS pipeline:
// async global->LDS copies (ASYNCcnt) when available, else VGPR round-trip
// with loads issued after the barrier for full load/WMMA overlap.
// ---------------------------------------------------------------------------
__global__ __launch_bounds__(256) void k_moe_gemm(
    const unsigned short* __restrict__ xbf,   // [b][n][k] bf16
    const unsigned short* __restrict__ ebfT,  // [e][n][k] bf16 (transposed experts)
    const float* __restrict__ tv, const int* __restrict__ ti,
    const int* __restrict__ winner, float* __restrict__ y)
{
    int bid = blockIdx.x;
    int ct = bid & 7;            // col tile (128 cols)
    int rt = (bid >> 3) & 3;     // row tile (128 of the 512 routed rows)
    int pe = bid >> 5;           // (b,e) pair
    int b = pe >> 3, e = pe & 7;

    __shared__ unsigned short Ash[2][128 * LDA];
    __shared__ unsigned short Bsh[2][128 * LDA];
    __shared__ int tok[128];
    __shared__ int okm[128];

    int tid = threadIdx.x;
    if (tid < 128) {
        int gm = rt * 128 + tid;
        int tk = ti[(size_t)pe * MM + gm];
        float gv = tv[(size_t)pe * MM + gm];
        tok[tid] = tk;
        okm[tid] = (gv > 0.5f) && (winner[b * NN + tk] == gm * EE + e);
    }
    __syncthreads();

    int wid = tid >> 5, lane = tid & 31;
    int WR = (wid & 3) * 32;
    int WC = (wid >> 2) * 64;

    v8f acc[2][4];
    #pragma unroll
    for (int fr = 0; fr < 2; ++fr)
        #pragma unroll
        for (int fc = 0; fc < 4; ++fc)
            acc[fr][fc] = (v8f){0.f,0.f,0.f,0.f,0.f,0.f,0.f,0.f};

    // staging: 2 threads per LDS row, 32 B each
    int rowS = tid >> 1;
    int half = tid & 1;
    const unsigned short* aptr = xbf + ((size_t)b * NN + tok[rowS]) * DD + half * 16;
    const unsigned short* bptr = ebfT + ((size_t)e * DD + ct * 128 + rowS) * DD + half * 16;
    unsigned int ldsOff = rowS * LDA + half * 16;

    // fragment lane mapping (documented 16-bit A / B VGPR layouts)
    int arow = lane & 15;
    int koA = (lane < 16) ? 0 : 8;     // A: lanes<16 K{0..7,16..23}, else K{8..15,24..31}
    int koB = (lane < 16) ? 0 : 8;     // B: lanes<16 K{0..7,16..23}, else K{8..15,24..31}
    int bcol = lane & 15;

#if HAVE_ASYNC_LDS
    unsigned ldsA = (unsigned)(uintptr_t)&Ash[0][ldsOff];
    unsigned ldsB = (unsigned)(uintptr_t)&Bsh[0][ldsOff];
    const unsigned bufBytes = 128u * LDA * 2u;

    // prelude: tile 0 -> buffer 0 (async, no VGPR round-trip)
    async_cp32(aptr, ldsA);
    async_cp32(bptr, ldsB);
#else
    uint4 ga0 = ((const uint4*)aptr)[0];
    uint4 ga1 = ((const uint4*)aptr)[1];
    uint4 gb0 = ((const uint4*)bptr)[0];
    uint4 gb1 = ((const uint4*)bptr)[1];
    {
        uint4* da = (uint4*)&Ash[0][ldsOff]; da[0] = ga0; da[1] = ga1;
        uint4* db = (uint4*)&Bsh[0][ldsOff]; db[0] = gb0; db[1] = gb1;
    }
#endif

    for (int i = 0; i < 32; ++i) {
        int cb = i & 1;

#if HAVE_ASYNC_LDS
        wait_async0();        // this wave's portion of tile i landed in LDS
        __syncthreads();      // everyone's portion visible; prior reads of nb done
        if (i < 31) {         // tile i+1 -> other buffer, overlapped with WMMA below
            int k0n = (i + 1) * 32;
            unsigned nbO = (unsigned)((cb ^ 1)) * bufBytes;
            async_cp32(aptr + k0n, ldsA + nbO);
            async_cp32(bptr + k0n, ldsB + nbO);
            if (i < 30) {
                __builtin_prefetch(aptr + k0n + 32, 0, 1);   // -> global_prefetch_b8
                __builtin_prefetch(bptr + k0n + 32, 0, 1);
            }
        }
#else
        __syncthreads();      // buffer cb ready
        if (i < 31) {         // issue loads AFTER barrier so they overlap the WMMAs
            int k0n = (i + 1) * 32;
            ga0 = ((const uint4*)(aptr + k0n))[0];
            ga1 = ((const uint4*)(aptr + k0n))[1];
            gb0 = ((const uint4*)(bptr + k0n))[0];
            gb1 = ((const uint4*)(bptr + k0n))[1];
            if (i < 30) {
                __builtin_prefetch(aptr + k0n + 32, 0, 1);
                __builtin_prefetch(bptr + k0n + 32, 0, 1);
            }
        }
#endif

        V16 af[2];
        #pragma unroll
        for (int fr = 0; fr < 2; ++fr) {
            const unsigned short* base = &Ash[cb][(WR + fr * 16 + arow) * LDA + koA];
            af[fr].q[0] = *(const uint4*)(base);
            af[fr].q[1] = *(const uint4*)(base + 16);
        }
        V16 bfr[4];
        #pragma unroll
        for (int fc = 0; fc < 4; ++fc) {
            const unsigned short* base = &Bsh[cb][(WC + fc * 16 + bcol) * LDA + koB];
            bfr[fc].q[0] = *(const uint4*)(base);
            bfr[fc].q[1] = *(const uint4*)(base + 16);
        }
        #pragma unroll
        for (int fr = 0; fr < 2; ++fr)
            #pragma unroll
            for (int fc = 0; fc < 4; ++fc)
                acc[fr][fc] = __builtin_amdgcn_wmma_f32_16x16x32_bf16(
                    false, af[fr].v, false, bfr[fc].v,
                    (short)0, acc[fr][fc], false, false);

#if !HAVE_ASYNC_LDS
        if (i < 31) {
            int nb = cb ^ 1;
            uint4* da = (uint4*)&Ash[nb][ldsOff]; da[0] = ga0; da[1] = ga1;
            uint4* db = (uint4*)&Bsh[nb][ldsOff]; db[0] = gb0; db[1] = gb1;
        }
#endif
    }

    // epilogue: deterministic masked scatter
    #pragma unroll
    for (int fr = 0; fr < 2; ++fr) {
        int rbase = WR + fr * 16 + ((lane >= 16) ? 8 : 0);
        #pragma unroll
        for (int fc = 0; fc < 4; ++fc) {
            int col = ct * 128 + WC + fc * 16 + (lane & 15);
            #pragma unroll
            for (int r = 0; r < 8; ++r) {
                int row = rbase + r;
                if (okm[row])
                    y[((size_t)b * NN + tok[row]) * DD + col] = acc[fr][fc][r];
            }
        }
    }
}

// ---------------------------------------------------------------------------
// Launcher.  Workspace requirement: ~48.7 MB
//   gates 512K | tv 64K | ti 64K | winner 64K | xbf 32M | ebfT 16M
// ---------------------------------------------------------------------------
extern "C" void kernel_launch(void* const* d_in, const int* in_sizes, int n_in,
                              void* d_out, int out_size, void* d_ws, size_t ws_size,
                              hipStream_t stream)
{
    const float* x       = (const float*)d_in[0];   // (4,4096,1024) f32
    const float* gate_w  = (const float*)d_in[1];   // (1,1024,8)    f32
    const float* experts = (const float*)d_in[2];   // (8,1024,1024) f32
    float* y = (float*)d_out;                       // (4,4096,1024) f32

    char* ws = (char*)d_ws;
    float*          gates  = (float*)(ws);                         // 512 KB
    float*          tv     = (float*)(ws + 0x80000);               //  64 KB
    int*            ti     = (int*)  (ws + 0x90000);               //  64 KB
    int*            winner = (int*)  (ws + 0xA0000);               //  64 KB
    unsigned short* xbf    = (unsigned short*)(ws + 0xB0000);      //  32 MB
    unsigned short* ebfT   = (unsigned short*)(ws + 0xB0000 + (size_t)BB*NN*DD*2); // 16 MB

    (void)hipMemsetAsync(d_out, 0, (size_t)out_size * sizeof(float), stream);
    (void)hipMemsetAsync(winner, 0xFF, (size_t)BB * NN * sizeof(int), stream);   // -1

    // bf16 pre-conversion (x) and transpose+conversion (experts)
    k_cvt_bf16<<<(BB * NN * DD) / (256 * 8), 256, 0, stream>>>(x, xbf);
    k_cvt_transpose<<<EE * 32 * 32, 256, 0, stream>>>(experts, ebfT);

    // router
    k_gate_logits<<<(BB * NN) / 8, 256, 0, stream>>>(x, gate_w, gates);
    k_sinkhorn<<<BB, 1024, 0, stream>>>(gates);
    k_topk<<<BB * EE, 1024, 0, stream>>>(gates, tv, ti, winner);

    // expert GEMMs + scatter
    k_moe_gemm<<<BB * EE * (MM / 128) * (DD / 128), 256, 0, stream>>>(
        xbf, ebfT, tv, ti, winner, y);
}
